// OutputBlock_52716428591196
// MI455X (gfx1250) — matrix operand order repaired and checked by reading, not compiled, not run on hardware
//
#include <hip/hip_runtime.h>
#include <hip/hip_bf16.h>

typedef __attribute__((ext_vector_type(16))) _Float16 v16h;
typedef __attribute__((ext_vector_type(8)))  _Float16 v8h;
typedef __attribute__((ext_vector_type(8)))  float    v8f;
typedef __attribute__((ext_vector_type(4)))  unsigned int v4u;
typedef __attribute__((ext_vector_type(8)))  int v8i;
typedef __attribute__((ext_vector_type(4)))  int v4i;

#define CIN 64
#define HH  256
#define WW  256

__device__ __forceinline__ float fq7(float v, float s) {
    // 7-bit symmetric fake-quant; rintf = round-half-even, matches jnp.round
    float q = rintf(v / s);
    q = fminf(fmaxf(q, -64.f), 63.f);
    return q * s;
}

// ---------------------------------------------------------------------------
// Pack fake-quantized weights into WMMA B-fragment layout (f16):
//   wq1f: [tap(9)][cb(2)][nt(4)][lane(32)][half(16)]
//   wq2f: [tap(9)][cb(2)][lane(32)][half(16)]   (out-channels 3..15 zero)
// B (K x N): lane L -> N = L%16, K = (L/16)*8 + (j%8) + (j/8)*16   (j = half index 0..15)
// ---------------------------------------------------------------------------
__global__ void pack_weights_kernel(const float* __restrict__ w1, const float* __restrict__ sw1,
                                    const float* __restrict__ w2, const float* __restrict__ sw2,
                                    _Float16* __restrict__ wq1f, _Float16* __restrict__ wq2f) {
    int idx = blockIdx.x * blockDim.x + threadIdx.x;
    if (idx < 9*2*4*32*16) {
        int j  = idx & 15;
        int L  = (idx >> 4) & 31;
        int nt = (idx >> 9) & 3;
        int cb = (idx >> 11) & 1;
        int t  = idx >> 12;
        int K  = ((L >> 4) << 3) + (j & 7) + ((j >> 3) << 4);
        int c  = cb * 32 + K;
        int oc = nt * 16 + (L & 15);
        float w = w1[((oc * CIN + c) * 3 + t / 3) * 3 + (t % 3)];
        float s = sw1[oc];
        float q = fminf(fmaxf(rintf(w / s), -64.f), 63.f);
        wq1f[idx] = (_Float16)(q * s);
    }
    int idx2 = idx - 9*2*4*32*16;
    if (idx2 >= 0 && idx2 < 9*2*32*16) {
        int j  = idx2 & 15;
        int L  = (idx2 >> 4) & 31;
        int cb = (idx2 >> 9) & 1;
        int t  = idx2 >> 10;
        int K  = ((L >> 4) << 3) + (j & 7) + ((j >> 3) << 4);
        int c  = cb * 32 + K;
        int oc = L & 15;
        float val = 0.f;
        if (oc < 3) {
            float w = w2[((oc * CIN + c) * 3 + t / 3) * 3 + (t % 3)];
            float s = sw2[oc];
            float q = fminf(fmaxf(rintf(w / s), -64.f), 63.f);
            val = q * s;
        }
        wq2f[idx2] = (_Float16)val;
    }
}

// ---------------------------------------------------------------------------
// conv1: implicit GEMM. Per wave: 2 M-tiles (2 rows x 16 px) x 64 out-ch so
// each B-fragment feeds 2 WMMAs (LDS traffic 1.5 b128/WMMA instead of 2.5).
// Workgroup = 16 rows x 16 cols. Per-tap weight staging keeps LDS < 64KB.
// ---------------------------------------------------------------------------
__global__ void __launch_bounds__(256) conv1_kernel(
    const float* __restrict__ x, const _Float16* __restrict__ wq1f,
    const float* __restrict__ b1, const float* __restrict__ alpha1,
    const float* __restrict__ sa1, _Float16* __restrict__ hbuf) {

    __shared__ _Float16 xs[18 * 18 * 64];   // 41472 B, layout [yy][xx][c]
    __shared__ _Float16 wl[8 * 512];        //  8192 B, one tap (2cb x 4nt frags)

    const int tid = threadIdx.x;
    const int bid = blockIdx.x;
    const int tx = bid & 15;
    const int ty = (bid >> 4) & 15;
    const int n  = bid >> 8;
    const int x0 = tx * 16;
    const int y0 = ty * 16;

    // stage input tile (f32 -> f16), zero-padded halo
    for (int i = tid; i < 18 * 18 * 64; i += 256) {
        int c  = i / 324;
        int r  = i - c * 324;
        int yy = r / 18;
        int xx = r - yy * 18;
        int gy = y0 - 1 + yy;
        int gx = x0 - 1 + xx;
        float v = 0.f;
        if ((unsigned)gy < 256u && (unsigned)gx < 256u)
            v = x[(((size_t)n * CIN + c) * HH + gy) * WW + gx];
        xs[(yy * 18 + xx) * 64 + c] = (_Float16)v;
    }

    const int lane = tid & 31;
    const int wave = tid >> 5;
    const int lm = lane & 15;   // A row M / D column N sub-index
    const int lh = lane >> 4;   // K-half select for A
    const int r0 = 2 * wave;    // first of this wave's two rows

    v8f acc[2][4] = {{v8f{}, v8f{}, v8f{}, v8f{}}, {v8f{}, v8f{}, v8f{}, v8f{}}};

    for (int t = 0; t < 9; ++t) {
        __syncthreads();  // xs ready / previous tap's weight reads done
        {   // stage this tap's 8 fragments: 512 x 16B chunks from L2
            const uint4* src = (const uint4*)wq1f + t * 512;
            uint4* dst = (uint4*)wl;
            dst[tid]       = src[tid];
            dst[tid + 256] = src[tid + 256];
        }
        __syncthreads();

        const int dy  = t / 3;
        const int dxx = t % 3;
        #pragma unroll
        for (int cb = 0; cb < 2; ++cb) {
            const int ab0 = ((r0 + dy) * 18 + lm + dxx) * 64 + cb * 32 + lh * 8;
            union { v16h v; v8h h[2]; } A0, A1;
            A0.h[0] = *(const v8h*)&xs[ab0];
            A0.h[1] = *(const v8h*)&xs[ab0 + 16];
            A1.h[0] = *(const v8h*)&xs[ab0 + 18 * 64];
            A1.h[1] = *(const v8h*)&xs[ab0 + 18 * 64 + 16];
            const _Float16* wb = &wl[cb * 4 * 512 + lane * 16];
            #pragma unroll
            for (int nt = 0; nt < 4; ++nt) {
                union { v16h v; v8h h[2]; } B;
                B.h[0] = *(const v8h*)&wb[nt * 512];
                B.h[1] = *(const v8h*)&wb[nt * 512 + 8];
                acc[0][nt] = __builtin_amdgcn_wmma_f32_16x16x32_f16(
                    false, A0.v, false, B.v, (short)0, acc[0][nt], false, false);
                acc[1][nt] = __builtin_amdgcn_wmma_f32_16x16x32_f16(
                    false, A1.v, false, B.v, (short)0, acc[1][nt], false, false);
            }
        }
    }

    // epilogue: bias + PReLU + act fake-quant -> NHWC f16 intermediate
    #pragma unroll
    for (int mt = 0; mt < 2; ++mt) {
        const int y = y0 + r0 + mt;
        #pragma unroll
        for (int nt = 0; nt < 4; ++nt) {
            const int ch = nt * 16 + lm;  // one out-channel per lane (D layout)
            const float bb = b1[ch], al = alpha1[ch], s = sa1[ch];
            #pragma unroll
            for (int r = 0; r < 8; ++r) {
                float v = acc[mt][nt][r] + bb;
                v = v >= 0.f ? v : al * v;
                v = fq7(v, s);
                const int col = x0 + r + lh * 8;  // M = r (lanes<16) or r+8
                hbuf[(((size_t)n * HH + y) * WW + col) * 64 + ch] = (_Float16)v;
            }
        }
    }
}

// ---------------------------------------------------------------------------
// conv2: 64 -> 3 (padded to 16). Intermediate is f16 NHWC, so the LDS tile
// fill is a pure 2D strided copy -> Tensor Data Mover (tensor_load_to_lds).
// D# uses iterate mode: 1-row inner tile, lds_addr_increment = LDS pitch,
// global_addr_increment = NHWC row pitch; boundary tiles clamp the rectangle
// and pre-zero LDS. 2 M-tiles per wave as in conv1.
// ---------------------------------------------------------------------------
__global__ void __launch_bounds__(256) conv2_kernel(
    const _Float16* __restrict__ hbuf, const _Float16* __restrict__ wq2f,
    const float* __restrict__ b2, const float* __restrict__ alpha2,
    const float* __restrict__ sa2, float* __restrict__ out) {

    __shared__ _Float16 hs[18 * 18 * 64];   // 41472 B
    __shared__ _Float16 wl[18 * 512];       // 18432 B

    const int tid = threadIdx.x;
    const int bid = blockIdx.x;
    const int tx = bid & 15;
    const int ty = (bid >> 4) & 15;
    const int n  = bid >> 8;
    const int x0 = tx * 16;
    const int y0 = ty * 16;

    {   // weights: 9216 halves = 1152 x 16B chunks (L2-resident)
        const uint4* src = (const uint4*)wq2f;
        uint4* dst = (uint4*)wl;
        for (int i = tid; i < 1152; i += 256) dst[i] = src[i];
    }
    // pre-zero tile only when the halo is clamped (edge workgroups)
    if (tx == 0 || tx == 15 || ty == 0 || ty == 15) {
        uint4 z = uint4{0u, 0u, 0u, 0u};
        for (int i = tid; i < 2592; i += 256) ((uint4*)hs)[i] = z;
    }
    __syncthreads();

    if ((tid >> 5) == 0) {   // wave 0 issues the TDM (EXEC-independent)
        const int cy0 = max(y0 - 1, 0), cy1 = min(y0 + 17, 256);
        const int cx0 = max(x0 - 1, 0), cx1 = min(x0 + 17, 256);
        const int nrows = cy1 - cy0;
        const int ncols = cx1 - cx0;
        const unsigned long long gaddr = (unsigned long long)(uintptr_t)hbuf +
            ((((unsigned long long)n * HH + cy0) * WW) + cx0) * 64ull * 2ull;
        const unsigned int laddr = (unsigned int)(uintptr_t)
            &hs[((cy0 - (y0 - 1)) * 18 + (cx0 - (x0 - 1))) * 64];

        v4u g0;                                   // D# group 0 (128b)
        g0[0] = 1u;                               // count=1, no gather
        g0[1] = laddr;                            // lds_addr (bytes)
        g0[2] = (unsigned int)gaddr;              // global_addr[31:0]
        g0[3] = (unsigned int)(gaddr >> 32) | (2u << 30);  // ga[56:32] | type=2

        v8i g1;                                   // D# group 1 (256b)
        g1[0] = (1 << 16) | (1 << 19);            // data_size=2B, iterate_enable
        g1[1] = (16384 << 16);                    // tensor_dim0 = 16384 (row elems)
        g1[2] = (1 << 16);                        // tensor_dim1 = 1 (per iteration)
        g1[3] = ((ncols * 64) << 16);             // tile_dim0 = ncols*64 elems
        g1[4] = 1;                                // tile_dim1 = 1, tile_dim2 = 0
        g1[5] = 16384;                            // tensor_dim0_stride lo32
        g1[6] = 0;
        g1[7] = 0;

        v4i g2;                                   // D# group 2: iterate fields
        g2[0] = 0;                                // tensor_dim2 unused
        g2[1] = 18 * 64;                          // lds_addr_increment (elements)
        g2[2] = 16384;                            // global_addr_increment lo32
        g2[3] = ((nrows - 1) << 16);              // iterate_count in [31:16]

        v4i g3 = v4i{0, 0, 0, 0};

#if __clang_major__ >= 23
        v8i gpad = v8i{0, 0, 0, 0, 0, 0, 0, 0};
        __builtin_amdgcn_tensor_load_to_lds(g0, g1, g2, g3, gpad, 0);
#else
        __builtin_amdgcn_tensor_load_to_lds(g0, g1, g2, g3, 0);
#endif
        __builtin_amdgcn_s_wait_tensorcnt(0);
    }
    __syncthreads();

    const int lane = tid & 31;
    const int wave = tid >> 5;
    const int lm = lane & 15;
    const int lh = lane >> 4;
    const int r0 = 2 * wave;

    v8f acc[2] = {v8f{}, v8f{}};
    #pragma unroll 3
    for (int t = 0; t < 9; ++t) {
        const int dy  = t / 3;
        const int dxx = t % 3;
        #pragma unroll
        for (int cb = 0; cb < 2; ++cb) {
            const int ab0 = ((r0 + dy) * 18 + lm + dxx) * 64 + cb * 32 + lh * 8;
            union { v16h v; v8h h[2]; } A0, A1;
            A0.h[0] = *(const v8h*)&hs[ab0];
            A0.h[1] = *(const v8h*)&hs[ab0 + 16];
            A1.h[0] = *(const v8h*)&hs[ab0 + 18 * 64];
            A1.h[1] = *(const v8h*)&hs[ab0 + 18 * 64 + 16];
            union { v16h v; v8h h[2]; } B;
            const _Float16* wb = &wl[(t * 2 + cb) * 512 + lane * 16];
            B.h[0] = *(const v8h*)&wb[0];
            B.h[1] = *(const v8h*)&wb[8];
            acc[0] = __builtin_amdgcn_wmma_f32_16x16x32_f16(
                false, A0.v, false, B.v, (short)0, acc[0], false, false);
            acc[1] = __builtin_amdgcn_wmma_f32_16x16x32_f16(
                false, A1.v, false, B.v, (short)0, acc[1], false, false);
        }
    }

    const int ch = lm;
    if (ch < 3) {   // divergence only after all WMMAs (EXEC all-1s during WMMA)
        const float bb = b2[ch], al = alpha2[ch], s = sa2[ch];
        #pragma unroll
        for (int mt = 0; mt < 2; ++mt) {
            const int y = y0 + r0 + mt;
            #pragma unroll
            for (int r = 0; r < 8; ++r) {
                float v = acc[mt][r] + bb;
                v = v >= 0.f ? v : al * v;
                v = fq7(v, s);
                const int col = x0 + r + lh * 8;
                out[(((size_t)n * 3 + ch) * HH + y) * WW + col] = v;
            }
        }
    }
}

// ---------------------------------------------------------------------------
extern "C" void kernel_launch(void* const* d_in, const int* in_sizes, int n_in,
                              void* d_out, int out_size, void* d_ws, size_t ws_size,
                              hipStream_t stream) {
    const float* x      = (const float*)d_in[0];
    const float* w1     = (const float*)d_in[1];
    const float* b1     = (const float*)d_in[2];
    const float* w2     = (const float*)d_in[3];
    const float* b2     = (const float*)d_in[4];
    const float* alpha1 = (const float*)d_in[5];
    const float* alpha2 = (const float*)d_in[6];
    const float* sw1    = (const float*)d_in[7];
    const float* sw2    = (const float*)d_in[8];
    const float* sa1    = (const float*)d_in[9];
    const float* sa2    = (const float*)d_in[10];
    float* out = (float*)d_out;

    // workspace: [wq1 frags 36864 halves][wq2 frags 9216 halves][h NHWC f16]
    _Float16* wq1f = (_Float16*)d_ws;
    _Float16* wq2f = wq1f + 36864;
    _Float16* hbuf = wq1f + 46080;   // byte offset 92160, 16B-aligned

    pack_weights_kernel<<<180, 256, 0, stream>>>(w1, sw1, w2, sw2, wq1f, wq2f);
    conv1_kernel<<<2048, 256, 0, stream>>>(x, wq1f, b1, alpha1, sa1, hbuf);
    conv2_kernel<<<2048, 256, 0, stream>>>(hbuf, wq2f, b2, alpha2, sa2, out);
}
